// DB_FC_CNN_36867999269166
// MI455X (gfx1250) — matrix operand order repaired and checked by reading, not compiled, not run on hardware
//
#include <hip/hip_runtime.h>
#include <math.h>

// ---------------- types ----------------
typedef _Float16 v16h __attribute__((ext_vector_type(16)));
typedef _Float16 v8h  __attribute__((ext_vector_type(8)));
typedef float    v8f  __attribute__((ext_vector_type(8)));

// ---------------- problem constants ----------------
constexpr int NB   = 128;    // batch
constexpr int NU   = 8;      // users
constexpr int NA   = 64;     // antennas
constexpr int HL   = 1024;   // 2*U*A sequence length
constexpr int HPAD = 1026;   // +1 halo row each side
constexpr float EPS = 1e-5f;

// =====================================================================
// Prep: pack conv weights (middle kw column) to f16 [Cout][3*Cin], k = kh*Cin+i
// =====================================================================
__global__ void k_prep_w(const float* __restrict__ W, _Float16* __restrict__ Wm,
                         int Cin, int Cout) {
    int idx = blockIdx.x * blockDim.x + threadIdx.x;
    int tot = Cout * Cin * 3;
    if (idx >= tot) return;
    int o  = idx / (Cin * 3);
    int r  = idx % (Cin * 3);
    int kh = r / Cin;
    int i  = r % Cin;
    Wm[(size_t)o * (3 * Cin) + kh * Cin + i] =
        (_Float16)W[(((size_t)o * Cin + i) * 3 + kh) * 3 + 1];
}

__global__ void k_prep_w4(const float* __restrict__ W, float* __restrict__ w4) {
    int idx = blockIdx.x * blockDim.x + threadIdx.x;
    if (idx >= 768) return;
    int kh = idx >> 8, i = idx & 255;
    w4[kh * 256 + i] = W[((size_t)i * 3 + kh) * 3 + 1];
}

// fold BN (eval) into per-channel scale/bias:  y = scale*conv + bias
__global__ void k_prep_sb(const float* g, const float* be, const float* mn,
                          const float* vr, const float* cb,
                          float* scale, float* bias, int C) {
    int c = blockIdx.x * blockDim.x + threadIdx.x;
    if (c >= C) return;
    float s = g[c] * rsqrtf(vr[c] + EPS);
    scale[c] = s;
    bias[c]  = (cb[c] - mn[c]) * s + be[c];
}

// zero the halo rows (h'=0 and h'=HPAD-1) of a [NB][HPAD][C] f16 buffer
__global__ void k_zero_halo(_Float16* __restrict__ x, int C) {
    int idx = blockIdx.x * blockDim.x + threadIdx.x;
    int tot = NB * 2 * C;
    if (idx >= tot) return;
    int b = idx / (2 * C);
    int r = idx % (2 * C);
    int row = (r < C) ? 0 : (HPAD - 1);
    int c   = r % C;
    x[((size_t)b * HPAD + row) * C + c] = (_Float16)0.f;
}

// =====================================================================
// Layer 1: conv(1->64,k3) + BN + ReLU, output channel-last f16 with halo
// =====================================================================
__global__ void k_layer1(const float* __restrict__ Hr, const float* __restrict__ Hi,
                         const float* __restrict__ W1, const float* __restrict__ cb,
                         const float* __restrict__ g,  const float* __restrict__ be,
                         const float* __restrict__ mn, const float* __restrict__ vr,
                         _Float16* __restrict__ x1t) {
    __shared__ float sw0[64], sw1[64], sw2[64], ssc[64], sbi[64];
    int t = threadIdx.x;
    if (t < 64) {
        float s = g[t] * rsqrtf(vr[t] + EPS);
        ssc[t] = s;
        sbi[t] = (cb[t] - mn[t]) * s + be[t];
        sw0[t] = W1[(t * 3 + 0) * 3 + 1];
        sw1[t] = W1[(t * 3 + 1) * 3 + 1];
        sw2[t] = W1[(t * 3 + 2) * 3 + 1];
    }
    __syncthreads();
    int idx = blockIdx.x * blockDim.x + t;     // [0, NB*1024)
    int b = idx >> 10, h = idx & 1023;
    auto x0 = [&](int j) -> float {
        if (j < 0 || j >= HL) return 0.f;
        return (j < 512) ? Hr[(size_t)b * 512 + j] : Hi[(size_t)b * 512 + j - 512];
    };
    float xm = x0(h - 1), xc = x0(h), xp = x0(h + 1);
    _Float16* dst = x1t + ((size_t)b * HPAD + h + 1) * 64;
    #pragma unroll 8
    for (int c = 0; c < 64; ++c) {
        float y = ssc[c] * (sw0[c] * xm + sw1[c] * xc + sw2[c] * xp) + sbi[c];
        dst[c] = (_Float16)(y > 0.f ? y : 0.f);
    }
}

// =====================================================================
// WMMA implicit-GEMM conv (k=3) + BN + ReLU, register-blocked 2x4 tiles.
//   xin : [NB][HPAD][CIN]  f16 (halo rows zero)
//   Wm  : [COUT][3*CIN]    f16, k = kh*CIN + i
//   xout: [NB][HPAD][COUT] f16 (interior rows written here)
// One wave computes MT(=2) cout-tiles x NT(=4) h-tiles of 16x16 each:
// per K-step, 2 A-frag + 4 B-frag loads feed 8 WMMAs (A reused 4x, B 2x).
// =====================================================================
template <int CIN, int COUT>
__global__ __launch_bounds__(128)
void k_conv_wmma(const _Float16* __restrict__ Wm,
                 const _Float16* __restrict__ xin,
                 _Float16* __restrict__ xout,
                 const float* __restrict__ scale,
                 const float* __restrict__ bias) {
    constexpr int KTOT = 3 * CIN;
    constexpr int MT = 2;                 // cout tiles per wave
    constexpr int NT = 4;                 // h tiles per wave
    constexpr int GM = COUT / 16 / MT;    // cout tile-groups
    constexpr int GN = (HL / 16) / NT;    // h tile-groups
    constexpr int TPB = GM * GN;          // wave-tiles per batch

    int gwave = (blockIdx.x * blockDim.x + threadIdx.x) >> 5;
    int lane  = threadIdx.x & 31;
    int b  = gwave / TPB;
    int r  = gwave % TPB;
    int og = r / GN;                      // cout group
    int hg = r % GN;                      // h group

    int m     = lane & 15;
    int hi16  = lane >> 4;     // 0 for lanes 0-15, 1 for 16-31
    int koffA = hi16 * 16;     // A frag: lanes 16-31 hold K+16..K+31
    int koffB = hi16 * 16;     // B frag: lanes 16-31 hold K+16..K+31

    const _Float16* wrow[MT];
    #pragma unroll
    for (int mi = 0; mi < MT; ++mi)
        wrow[mi] = Wm + (size_t)((og * MT + mi) * 16 + m) * KTOT + koffA;

    int hrow[NT];
    #pragma unroll
    for (int ni = 0; ni < NT; ++ni)
        hrow[ni] = (hg * NT + ni) * 16 + m;   // A row m == B col n == lane&15

    v8f acc[MT][NT];
    #pragma unroll
    for (int mi = 0; mi < MT; ++mi)
        #pragma unroll
        for (int ni = 0; ni < NT; ++ni)
            acc[mi][ni] = {};

    #pragma unroll
    for (int kb = 0; kb < KTOT / 32; ++kb) {
        int kbase = kb * 32;
        // ---- A fragments: 16x32 f16 weights, one per cout tile ----
        v16h a[MT];
        #pragma unroll
        for (int mi = 0; mi < MT; ++mi) {
            const _Float16* wp = wrow[mi] + kbase;
            v8h alo = *(const v8h*)(wp);        // K = kbase+koffA    .. +7
            v8h ahi = *(const v8h*)(wp + 8);    // K = kbase+8+koffA  .. +7
            a[mi] = __builtin_shufflevector(alo, ahi,
                        0,1,2,3,4,5,6,7,8,9,10,11,12,13,14,15);
        }
        // ---- B fragments: 32x16 f16 activations (im2col on the fly) ----
        int kstart = kbase + koffB;
        int kh = kstart / CIN;
        int i0 = kstart % CIN;              // 16 consecutive channels
        v16h bf[NT];
        #pragma unroll
        for (int ni = 0; ni < NT; ++ni)
            bf[ni] = *(const v16h*)(xin + ((size_t)b * HPAD + hrow[ni] + kh) * CIN + i0);
        // ---- 8 WMMAs, fragments reused from registers ----
        #pragma unroll
        for (int mi = 0; mi < MT; ++mi)
            #pragma unroll
            for (int ni = 0; ni < NT; ++ni)
                acc[mi][ni] = __builtin_amdgcn_wmma_f32_16x16x32_f16(
                                  false, a[mi], false, bf[ni], (short)0,
                                  acc[mi][ni], false, false);
    }

    // ---- fused BN + ReLU, store transposed into channel-last buffer ----
    #pragma unroll
    for (int mi = 0; mi < MT; ++mi) {
        int cbase = (og * MT + mi) * 16 + hi16 * 8;  // D: VGPR v -> M = v + hi16*8
        float sc8[8], bi8[8];
        #pragma unroll
        for (int v = 0; v < 8; ++v) { sc8[v] = scale[cbase + v]; bi8[v] = bias[cbase + v]; }
        #pragma unroll
        for (int ni = 0; ni < NT; ++ni) {
            v8h outv;
            #pragma unroll
            for (int v = 0; v < 8; ++v) {
                float y = acc[mi][ni][v] * sc8[v] + bi8[v];
                outv[v] = (_Float16)(y > 0.f ? y : 0.f);
            }
            *(v8h*)(xout + ((size_t)b * HPAD + hrow[ni] + 1) * COUT + cbase) = outv;
        }
    }
}

// =====================================================================
// Layer 4: conv(256->1,k3) + BN + ReLU -> feat [NB][1024] f32.
// One wave per (b,h); lane-strided dot over 768 taps + shuffle reduce.
// =====================================================================
__global__ void k_layer4(const float* __restrict__ w4,
                         const _Float16* __restrict__ x3t,
                         const float* __restrict__ g,  const float* __restrict__ be,
                         const float* __restrict__ mn, const float* __restrict__ vr,
                         const float* __restrict__ cb,
                         float* __restrict__ feat) {
    int wave = (blockIdx.x * blockDim.x + threadIdx.x) >> 5;
    int lane = threadIdx.x & 31;
    int b = wave >> 10, h = wave & 1023;
    float s = 0.f;
    #pragma unroll 4
    for (int j = lane; j < 768; j += 32) {
        int kh = j >> 8, i = j & 255;
        s += w4[j] * (float)x3t[((size_t)b * HPAD + h + kh) * 256 + i];
    }
    #pragma unroll
    for (int d = 16; d > 0; d >>= 1) s += __shfl_down(s, d, 32);
    if (lane == 0) {
        float sc = g[0] * rsqrtf(vr[0] + EPS);
        float y  = sc * (s + cb[0] - mn[0]) + be[0];
        feat[(size_t)b * HL + h] = y > 0.f ? y : 0.f;
    }
}

// =====================================================================
// MLP heads + per-batch column normalization -> p_pw, q_pw [NB][NU]
// =====================================================================
__global__ void k_mlp(const float* __restrict__ feat,
                      const float* __restrict__ pW, const float* __restrict__ pb,
                      const float* __restrict__ qW, const float* __restrict__ qb,
                      float* __restrict__ p_pw, float* __restrict__ q_pw) {
    __shared__ float D[16];   // [head*8 + u]
    int b = blockIdx.x, t = threadIdx.x;
    if (t < 16) {
        int head = t >> 3, u = t & 7;
        const float* Wrow = (head == 0 ? pW : qW) + (size_t)u * HL;
        const float* bb   = (head == 0 ? pb : qb);
        const float* f    = feat + (size_t)b * HL;
        float acc = bb[u];
        for (int j = 0; j < HL; ++j) acc += Wrow[j] * f[j];
        D[t] = 1.f / (1.f + expf(-acc));
    }
    __syncthreads();
    if (t < 16) {
        int head = t >> 3, u = t & 7;
        float ssum = 0.f;
        #pragma unroll
        for (int uu = 0; uu < 8; ++uu) ssum += D[head * 8 + uu];
        float val = 10.f * D[t] / ssum;
        (head == 0 ? p_pw : q_pw)[(size_t)b * NU + u] = val;
    }
}

// =====================================================================
// Per-batch: N = I + sum_u q_u h_u^H h_u ; X = N^{-1} H^H (Gauss-Jordan,
// HPD so no pivoting); V = (sqrt(p)/||X_col||) X ; T=|HV|^2 ; rate.
// 64 threads per batch (thread t owns row a=t).
// =====================================================================
__device__ inline float2 cmul(float2 a, float2 b) {
    return make_float2(a.x * b.x - a.y * b.y, a.x * b.y + a.y * b.x);
}

__global__ void k_solve(const float* __restrict__ Hr, const float* __restrict__ Hi,
                        const float* __restrict__ p_pw, const float* __restrict__ q_pw,
                        float* __restrict__ out) {
    __shared__ float2 Nm[64 * 64];   // 32 KB
    __shared__ float2 Xm[64 * 8];    //  4 KB
    __shared__ float2 Hs[8 * 64];    //  4 KB
    __shared__ float  qv[8], scv[8], Tm[64];

    int b = blockIdx.x, t = threadIdx.x;   // t in [0,64)
    // load H and q
    for (int u = 0; u < 8; ++u) {
        size_t off = ((size_t)b * NU + u) * NA + t;
        Hs[u * 64 + t] = make_float2(Hr[off], Hi[off]);
    }
    if (t < 8) qv[t] = q_pw[(size_t)b * NU + t];
    __syncthreads();

    // build N (Hermitian, row t) and RHS X = conj(H)^T
    for (int c = 0; c < 64; ++c) {
        float re = (t == c) ? 1.f : 0.f, im = 0.f;
        #pragma unroll
        for (int u = 0; u < 8; ++u) {
            float2 ha = Hs[u * 64 + t];   // conj(H[u][t])
            float2 hc = Hs[u * 64 + c];   // H[u][c]
            float q = qv[u];
            re += q * (ha.x * hc.x + ha.y * hc.y);
            im += q * (ha.x * hc.y - ha.y * hc.x);
        }
        Nm[t * 64 + c] = make_float2(re, im);
    }
    #pragma unroll
    for (int u = 0; u < 8; ++u) {
        float2 hh = Hs[u * 64 + t];
        Xm[t * 8 + u] = make_float2(hh.x, -hh.y);
    }

    // Gauss-Jordan elimination
    for (int k = 0; k < 64; ++k) {
        __syncthreads();
        float inv = 1.f / Nm[k * 64 + k].x;   // HPD -> real positive diagonal
        __syncthreads();
        Nm[k * 64 + t].x *= inv;              // scale row k (column-parallel)
        Nm[k * 64 + t].y *= inv;
        if (t < 8) { Xm[k * 8 + t].x *= inv; Xm[k * 8 + t].y *= inv; }
        __syncthreads();
        if (t != k) {
            float2 f = Nm[t * 64 + k];
            for (int c = 0; c < 64; ++c) {
                float2 p = cmul(f, Nm[k * 64 + c]);
                Nm[t * 64 + c].x -= p.x;
                Nm[t * 64 + c].y -= p.y;
            }
            #pragma unroll
            for (int u = 0; u < 8; ++u) {
                float2 p = cmul(f, Xm[k * 8 + u]);
                Xm[t * 8 + u].x -= p.x;
                Xm[t * 8 + u].y -= p.y;
            }
        }
    }
    __syncthreads();

    // column norms and scales
    if (t < 8) {
        float nr = 0.f;
        for (int a = 0; a < 64; ++a) {
            float2 x = Xm[a * 8 + t];
            nr += x.x * x.x + x.y * x.y;
        }
        scv[t] = sqrtf(p_pw[(size_t)b * NU + t]) / sqrtf(nr);
    }
    __syncthreads();

    // V output: interleaved complex64, [B][A][U]
    float2* Vout = (float2*)out;
    #pragma unroll
    for (int u = 0; u < 8; ++u) {
        float2 x = Xm[t * 8 + u];
        Vout[((size_t)b * NA + t) * NU + u] = make_float2(scv[u] * x.x, scv[u] * x.y);
    }

    // T[u][k] = |sum_a H[u][a] V[a][k]|^2 (thread t -> (u,k))
    {
        int u = t >> 3, k2 = t & 7;
        float re = 0.f, im = 0.f;
        for (int a = 0; a < 64; ++a) {
            float2 hh = Hs[u * 64 + a];
            float2 xx = Xm[a * 8 + k2];
            float vx = scv[k2] * xx.x, vy = scv[k2] * xx.y;
            re += hh.x * vx - hh.y * vy;
            im += hh.x * vy + hh.y * vx;
        }
        Tm[t] = re * re + im * im;
    }
    __syncthreads();
    if (t == 0) {
        float rate = 0.f;
        #pragma unroll
        for (int u = 0; u < 8; ++u) {
            float s = Tm[u * 8 + u], tot = 0.f;
            #pragma unroll
            for (int k2 = 0; k2 < 8; ++k2) tot += Tm[u * 8 + k2];
            rate += log2f(1.f + s / (tot - s + 1.f));
        }
        out[(size_t)NB * NA * NU * 2 + b] = rate;   // after V block
    }
}

// =====================================================================
// Host launcher
// =====================================================================
extern "C" void kernel_launch(void* const* d_in, const int* in_sizes, int n_in,
                              void* d_out, int out_size, void* d_ws, size_t ws_size,
                              hipStream_t stream) {
    (void)in_sizes; (void)n_in; (void)out_size; (void)ws_size;

    const float* Hr = (const float*)d_in[0];
    const float* Hi = (const float*)d_in[1];
    // conv params: layer l -> base 2 + 6*l : W, b, gamma, beta, mean, var
    auto CP = [&](int l, int j) -> const float* { return (const float*)d_in[2 + 6 * l + j]; };
    const float* pW = (const float*)d_in[26];
    const float* pb = (const float*)d_in[27];
    const float* qW = (const float*)d_in[28];
    const float* qb = (const float*)d_in[29];
    float* out = (float*)d_out;

    // ---- workspace carve-up ----
    char* ws = (char*)d_ws;
    size_t off = 0;
    auto alloc = [&](size_t bytes) -> char* {
        char* p = ws + off;
        off = (off + bytes + 255) & ~(size_t)255;
        return p;
    };
    _Float16* x1t  = (_Float16*)alloc((size_t)NB * HPAD * 64  * 2);
    _Float16* x2t  = (_Float16*)alloc((size_t)NB * HPAD * 128 * 2);
    _Float16* x3t  = (_Float16*)alloc((size_t)NB * HPAD * 256 * 2);
    float*    feat = (float*)   alloc((size_t)NB * HL * 4);
    _Float16* Wm2  = (_Float16*)alloc((size_t)128 * 192 * 2);
    _Float16* Wm3  = (_Float16*)alloc((size_t)256 * 384 * 2);
    float*    w4   = (float*)   alloc(768 * 4);
    float*    sc2  = (float*)   alloc(128 * 4);
    float*    bi2  = (float*)   alloc(128 * 4);
    float*    sc3  = (float*)   alloc(256 * 4);
    float*    bi3  = (float*)   alloc(256 * 4);
    float*    p_pw = (float*)   alloc((size_t)NB * NU * 4);
    float*    q_pw = (float*)   alloc((size_t)NB * NU * 4);

    // ---- weight / BN prep ----
    k_prep_w<<<(128 * 64 * 3 + 255) / 256, 256, 0, stream>>>(CP(1, 0), Wm2, 64, 128);
    k_prep_w<<<(256 * 128 * 3 + 255) / 256, 256, 0, stream>>>(CP(2, 0), Wm3, 128, 256);
    k_prep_w4<<<3, 256, 0, stream>>>(CP(3, 0), w4);
    k_prep_sb<<<1, 128, 0, stream>>>(CP(1, 2), CP(1, 3), CP(1, 4), CP(1, 5), CP(1, 1), sc2, bi2, 128);
    k_prep_sb<<<1, 256, 0, stream>>>(CP(2, 2), CP(2, 3), CP(2, 4), CP(2, 5), CP(2, 1), sc3, bi3, 256);

    // ---- zero halos ----
    k_zero_halo<<<(NB * 2 * 64  + 255) / 256, 256, 0, stream>>>(x1t, 64);
    k_zero_halo<<<(NB * 2 * 128 + 255) / 256, 256, 0, stream>>>(x2t, 128);
    k_zero_halo<<<(NB * 2 * 256 + 255) / 256, 256, 0, stream>>>(x3t, 256);

    // ---- CNN backbone ----
    k_layer1<<<NB * HL / 256, 256, 0, stream>>>(Hr, Hi, CP(0, 0), CP(0, 1),
                                                CP(0, 2), CP(0, 3), CP(0, 4), CP(0, 5), x1t);
    // layer2: 128x192 @ 192x(128*1024); 8192 waves, each 2x4 tiles of 16x16
    k_conv_wmma<64, 128><<<NB * (128 / 16 / 2) * (HL / 16 / 4) / 4, 128, 0, stream>>>(Wm2, x1t, x2t, sc2, bi2);
    // layer3: 256x384 @ 384x(128*1024); 16384 waves
    k_conv_wmma<128, 256><<<NB * (256 / 16 / 2) * (HL / 16 / 4) / 4, 128, 0, stream>>>(Wm3, x2t, x3t, sc3, bi3);
    k_layer4<<<NB * HL / 8, 256, 0, stream>>>(w4, x3t, CP(3, 2), CP(3, 3), CP(3, 4), CP(3, 5), CP(3, 1), feat);

    // ---- heads ----
    k_mlp<<<NB, 32, 0, stream>>>(feat, pW, pb, qW, qb, p_pw, q_pw);

    // ---- beamforming solve + rate ----
    k_solve<<<NB, 64, 0, stream>>>(Hr, Hi, p_pw, q_pw, out);
}
